// MultiHeadedAttention_12206297055623
// MI455X (gfx1250) — matrix operand order repaired and checked by reading, not compiled, not run on hardware
//
#include <hip/hip_runtime.h>
#include <hip/hip_bf16.h>

// MI455X (gfx1250) WMMA implementation of GQA multi-head attention.
// All GEMM stages use v_wmma_f32_16x16x32_f16 (f16 operands, fp32 accumulate).

#define Bn   4
#define Tn   2048
#define En   2048
#define NHn  16
#define NKVn 4
#define HDn  128

typedef __attribute__((ext_vector_type(16))) _Float16 v16h;
typedef __attribute__((ext_vector_type(8)))  float    v8f;

// ---------------------------------------------------------------------------
// Fragment loaders (layouts per CDNA5 ISA 7.12.2)
// A 16x32 f16: lane<16 -> row=lane,    halves = K[0..7],  K[16..23]
//              lane>=16 -> row=lane-16, halves = K[8..15], K[24..31]
// B 32x16 f16: lane<16 -> col=lane,    halves = K[0..7],  K[16..23]
//              lane>=16 -> col=lane-16, halves = K[8..15], K[24..31]
// C/D 16x16 f32: vgpr i, lane -> (M = i + 8*(lane>=16), N = lane&15)
// ---------------------------------------------------------------------------
__device__ __forceinline__ v16h frag_a_h(const _Float16* p0, int half) {
  const _Float16* p = p0 + 8 * half;
  union { v16h v; float4 f[2]; } u;
  u.f[0] = *reinterpret_cast<const float4*>(p);
  u.f[1] = *reinterpret_cast<const float4*>(p + 16);
  return u.v;
}
__device__ __forceinline__ v16h frag_a_f(const float* p0, int half) {
  const float* p = p0 + 8 * half;
  float4 a = *reinterpret_cast<const float4*>(p);
  float4 b = *reinterpret_cast<const float4*>(p + 4);
  float4 c = *reinterpret_cast<const float4*>(p + 16);
  float4 d = *reinterpret_cast<const float4*>(p + 20);
  v16h v;
  v[0]  = (_Float16)a.x; v[1]  = (_Float16)a.y; v[2]  = (_Float16)a.z; v[3]  = (_Float16)a.w;
  v[4]  = (_Float16)b.x; v[5]  = (_Float16)b.y; v[6]  = (_Float16)b.z; v[7]  = (_Float16)b.w;
  v[8]  = (_Float16)c.x; v[9]  = (_Float16)c.y; v[10] = (_Float16)c.z; v[11] = (_Float16)c.w;
  v[12] = (_Float16)d.x; v[13] = (_Float16)d.y; v[14] = (_Float16)d.z; v[15] = (_Float16)d.w;
  return v;
}
__device__ __forceinline__ v16h frag_a(const _Float16* p, int h) { return frag_a_h(p, h); }
__device__ __forceinline__ v16h frag_a(const float* p, int h)    { return frag_a_f(p, h); }

__device__ __forceinline__ v16h frag_b_h(const _Float16* p0, int half) {
  const _Float16* p = p0 + 8 * half;
  union { v16h v; float4 f[2]; } u;
  u.f[0] = *reinterpret_cast<const float4*>(p);
  u.f[1] = *reinterpret_cast<const float4*>(p + 16);
  return u.v;
}

__device__ __forceinline__ v8f wmma_f16(v16h a, v16h b, v8f c) {
  return __builtin_amdgcn_wmma_f32_16x16x32_f16(false, a, false, b, (short)0, c,
                                                false, false);
}

// ---------------------------------------------------------------------------
// Weight transpose + f32 -> f16 convert:  W (K x N) -> Wt (N x K)
// ---------------------------------------------------------------------------
__global__ void wtrans_kernel(const float* __restrict__ W, _Float16* __restrict__ Wt,
                              int K, int N) {
  long idx = (long)blockIdx.x * blockDim.x + threadIdx.x;
  long total = (long)K * N;
  if (idx >= total) return;
  int k = (int)(idx / N);
  int n = (int)(idx % N);
  Wt[(long)n * K + k] = (_Float16)W[idx];
}

// ---------------------------------------------------------------------------
// WMMA GEMM:  C(MxN) = A(MxK) @ Bt(NxK)^T + bias ; wave computes 32x32 tile
// ---------------------------------------------------------------------------
template <typename AT, bool F32OUT>
__global__ __launch_bounds__(256) void gemm_wmma_kernel(
    const AT* __restrict__ A, const _Float16* __restrict__ Bt,
    const float* __restrict__ bias, void* __restrict__ Cout,
    int M, int N, int K) {
  int lane = threadIdx.x & 31;
  int wave = threadIdx.x >> 5;
  int half = lane >> 4;
  int lr   = lane & 15;

  long tile = (long)blockIdx.x * 8 + wave;
  int  tnc  = N >> 5;
  int  tm   = (int)(tile / tnc);
  int  tn   = (int)(tile % tnc);
  int  M0   = tm * 32, N0 = tn * 32;
  if (M0 >= M) return;   // wave-uniform: EXEC stays full for WMMA

  const AT*       Ar0 = A  + (long)(M0 + lr)      * K;
  const AT*       Ar1 = A  + (long)(M0 + 16 + lr) * K;
  const _Float16* Bc0 = Bt + (long)(N0 + lr)      * K;
  const _Float16* Bc1 = Bt + (long)(N0 + 16 + lr) * K;

  v8f z = {0.f, 0.f, 0.f, 0.f, 0.f, 0.f, 0.f, 0.f};
  v8f c00 = z, c01 = z, c10 = z, c11 = z;

  for (int k0 = 0; k0 < K; k0 += 32) {
    v16h a0 = frag_a(Ar0 + k0, half);
    v16h a1 = frag_a(Ar1 + k0, half);
    v16h b0 = frag_b_h(Bc0 + k0, half);
    v16h b1 = frag_b_h(Bc1 + k0, half);
    if (k0 + 32 < K) {               // global_prefetch_b8 of next K-slab
      __builtin_prefetch(Ar0 + k0 + 32);
      __builtin_prefetch(Bc0 + k0 + 32);
    }
    c00 = wmma_f16(a0, b0, c00);
    c01 = wmma_f16(a0, b1, c01);
    c10 = wmma_f16(a1, b0, c10);
    c11 = wmma_f16(a1, b1, c11);
  }

  float bi0 = bias[N0 + lr];
  float bi1 = bias[N0 + 16 + lr];
#pragma unroll
  for (int i = 0; i < 8; i++) {
    long r0 = M0 + i + 8 * half;
    long r1 = r0 + 16;
    if (F32OUT) {
      float* C = (float*)Cout;
      C[r0 * N + N0 + lr]      = c00[i] + bi0;
      C[r0 * N + N0 + 16 + lr] = c01[i] + bi1;
      C[r1 * N + N0 + lr]      = c10[i] + bi0;
      C[r1 * N + N0 + 16 + lr] = c11[i] + bi1;
    } else {
      _Float16* C = (_Float16*)Cout;
      C[r0 * N + N0 + lr]      = (_Float16)(c00[i] + bi0);
      C[r0 * N + N0 + 16 + lr] = (_Float16)(c01[i] + bi1);
      C[r1 * N + N0 + lr]      = (_Float16)(c10[i] + bi0);
      C[r1 * N + N0 + 16 + lr] = (_Float16)(c11[i] + bi1);
    }
  }
}

// ---------------------------------------------------------------------------
// RoPE on K and V (reference applies rope to BOTH), honoring the reference's
// reshape semantics: k/v[b,n,t,d] = proj[b, n*512 + t/4, (t%4)*128 + d].
// Writes Kh logical (b,n,t,d) and Vt transposed (b,n,d,t), both f16.
// ---------------------------------------------------------------------------
__global__ void rope_kv_kernel(const _Float16* __restrict__ Kp,
                               const _Float16* __restrict__ Vp,
                               _Float16* __restrict__ Kh,
                               _Float16* __restrict__ Vt) {
  long idx = (long)blockIdx.x * blockDim.x + threadIdx.x;
  long total = (long)Bn * NKVn * Tn * (HDn / 2);
  if (idx >= total) return;
  int  j  = (int)(idx & 63);           // pair index d/2
  long r  = idx >> 6;
  int  t  = (int)(r % Tn);
  long r2 = r / Tn;
  int  n  = (int)(r2 % NKVn);
  int  b  = (int)(r2 / NKVn);

  long src = ((long)(b * Tn + n * 512 + (t >> 2))) * 512 + (t & 3) * 128 + 2 * j;
  // theta_j = 10000^(-2j/128) = exp(-j * ln(10000)/64)
  float th = __expf(-(float)j * 0.14391156831212787f);
  float ang = (float)t * th;
  float s, c;
  sincosf(ang, &s, &c);

  float k1 = (float)Kp[src], k2 = (float)Kp[src + 1];
  long dK = ((long)((b * NKVn + n) * Tn + t)) * HDn + 2 * j;
  Kh[dK]     = (_Float16)(k1 * c - k2 * s);
  Kh[dK + 1] = (_Float16)(k1 * s + k2 * c);

  float v1 = (float)Vp[src], v2 = (float)Vp[src + 1];
  long dV = ((long)((b * NKVn + n) * HDn + 2 * j)) * Tn + t;
  Vt[dV]      = (_Float16)(v1 * c - v2 * s);
  Vt[dV + Tn] = (_Float16)(v1 * s + v2 * c);
}

// ---------------------------------------------------------------------------
// Flash-style attention. Block = 4 waves = 64 q rows of one (b,h).
// Per wave: 16 q rows; key tiles of 32; online softmax; O accum 16x128 fp32.
// Reference quirks: scores *= sqrt(HD); masked -> 1e-9 (NOT -inf); kv = h%4.
// ---------------------------------------------------------------------------
__global__ __launch_bounds__(128) void attn_kernel(
    const _Float16* __restrict__ Qp,   // projection layout (B*T, 2048) f16
    const _Float16* __restrict__ Kh,   // (B,NKV,T,HD) f16, roped
    const _Float16* __restrict__ Vt,   // (B,NKV,HD,T) f16, roped, transposed
    const int* __restrict__ mask,      // (B,T,T) int32
    _Float16* __restrict__ Oh)         // (B*T, NH*HD) f16
{
  __shared__ _Float16 plds[4 * 16 * 32];   // 1KB per wave: P tile 16x32

  int lane = threadIdx.x & 31;
  int wave = threadIdx.x >> 5;
  int half = lane >> 4;
  int lr   = lane & 15;

  int qt = blockIdx.x & 31;            // T/64 = 32
  int h  = (blockIdx.x >> 5) & 15;     // NH
  int b  = blockIdx.x >> 9;            // B
  int kv = h & 3;                      // jnp.tile -> modulo
  int q0 = qt * 64 + wave * 16;

  // Q fragments, 4 K-steps over HD, through the reshape mapping:
  // q[b,h,t,d] = Qp[b*T + h*128 + t/16, (t%16)*128 + d]
  int tq = q0 + lr;
  const _Float16* qrow =
      Qp + ((long)(b * Tn) + h * 128 + (tq >> 4)) * (long)En + (tq & 15) * HDn;
  v16h qf[4];
#pragma unroll
  for (int ks = 0; ks < 4; ks++) qf[ks] = frag_a_h(qrow + ks * 32, half);

  const _Float16* Kbase = Kh + ((long)(b * NKVn + kv)) * Tn * HDn;
  const _Float16* Vbase = Vt + ((long)(b * NKVn + kv)) * HDn * Tn;
  const int*      mbase = mask + (long)b * Tn * Tn;

  v8f z = {0.f, 0.f, 0.f, 0.f, 0.f, 0.f, 0.f, 0.f};
  v8f o[8];
#pragma unroll
  for (int n = 0; n < 8; n++) o[n] = z;
  float m[8], l[8];
#pragma unroll
  for (int i = 0; i < 8; i++) { m[i] = -1e30f; l[i] = 0.f; }

  const float SC = 11.313708498984761f;   // sqrt(128): reference MULTIPLIES

  for (int t0 = 0; t0 < Tn; t0 += 32) {
    // S = Q @ K^T for 16 q rows x 32 keys
    v8f s0 = z, s1 = z;
#pragma unroll
    for (int ks = 0; ks < 4; ks++) {
      v16h kf0 = frag_b_h(Kbase + (long)(t0 + lr) * HDn + ks * 32, half);
      v16h kf1 = frag_b_h(Kbase + (long)(t0 + 16 + lr) * HDn + ks * 32, half);
      s0 = wmma_f16(qf[ks], kf0, s0);
      s1 = wmma_f16(qf[ks], kf1, s1);
    }

    float alpha[8];
#pragma unroll
    for (int i = 0; i < 8; i++) {
      int qr = q0 + i + 8 * half;
      const int* mp = mbase + (long)qr * Tn + t0 + lr;
      float v0 = s0[i] * SC;
      float v1 = s1[i] * SC;
      v0 = (mp[0]  == 0) ? 1e-9f : v0;   // branchless: EXEC stays full
      v1 = (mp[16] == 0) ? 1e-9f : v1;

      // row max across the 16 lanes of this half-wave
      float mx = fmaxf(v0, v1);
      mx = fmaxf(mx, __shfl_xor(mx, 1));
      mx = fmaxf(mx, __shfl_xor(mx, 2));
      mx = fmaxf(mx, __shfl_xor(mx, 4));
      mx = fmaxf(mx, __shfl_xor(mx, 8));
      float mn = fmaxf(m[i], mx);
      alpha[i] = __expf(m[i] - mn);
      m[i] = mn;

      v0 = __expf(v0 - mn);
      v1 = __expf(v1 - mn);
      float rs = v0 + v1;
      rs += __shfl_xor(rs, 1);
      rs += __shfl_xor(rs, 2);
      rs += __shfl_xor(rs, 4);
      rs += __shfl_xor(rs, 8);
      l[i] = l[i] * alpha[i] + rs;

      // C-layout P -> LDS row-major 16x32 f16 (for A-fragment reload)
      _Float16* pl = plds + wave * 512 + (i + 8 * half) * 32;
      pl[lr]      = (_Float16)v0;
      pl[lr + 16] = (_Float16)v1;
    }
    asm volatile("s_wait_dscnt 0" ::: "memory");
    v16h pf = frag_a_h(plds + wave * 512 + lr * 32, half);

    // O = O*alpha + P @ V   (V^T layout -> contiguous B-fragments over keys)
#pragma unroll
    for (int n = 0; n < 8; n++) {
#pragma unroll
      for (int i = 0; i < 8; i++) o[n][i] *= alpha[i];
      v16h vf = frag_b_h(Vbase + (long)(n * 16 + lr) * Tn + t0, half);
      o[n] = wmma_f16(pf, vf, o[n]);
    }
  }

  // Epilogue: normalize by row sums, store f16 to (b, t, h*HD + d)
#pragma unroll
  for (int i = 0; i < 8; i++) {
    float inv = 1.0f / l[i];
    int qr = q0 + i + 8 * half;
    _Float16* orow = Oh + ((long)(b * Tn + qr)) * En + h * HDn;
#pragma unroll
    for (int n = 0; n < 8; n++)
      orow[n * 16 + lr] = (_Float16)(o[n][i] * inv);
  }
}

// ---------------------------------------------------------------------------
// Host launch
// ---------------------------------------------------------------------------
extern "C" void kernel_launch(void* const* d_in, const int* in_sizes, int n_in,
                              void* d_out, int out_size, void* d_ws, size_t ws_size,
                              hipStream_t stream) {
  const float* x    = (const float*)d_in[0];
  const int*   mask = (const int*)  d_in[1];
  const float* Wq   = (const float*)d_in[2];
  const float* bq   = (const float*)d_in[3];
  const float* Wk   = (const float*)d_in[4];
  const float* bk   = (const float*)d_in[5];
  const float* Wv   = (const float*)d_in[6];
  const float* bv   = (const float*)d_in[7];
  const float* Wo   = (const float*)d_in[8];
  const float* bo   = (const float*)d_in[9];

  char* ws = (char*)d_ws;
  size_t off = 0;
  _Float16* Wq_t = (_Float16*)(ws + off); off += (size_t)En * En * 2;          // 8 MB
  _Float16* Wk_t = (_Float16*)(ws + off); off += (size_t)En * 512 * 2;         // 2 MB
  _Float16* Wv_t = (_Float16*)(ws + off); off += (size_t)En * 512 * 2;         // 2 MB
  _Float16* Wo_t = (_Float16*)(ws + off); off += (size_t)En * En * 2;          // 8 MB
  _Float16* Qp   = (_Float16*)(ws + off); off += (size_t)Bn * Tn * En * 2;     // 32 MB
  _Float16* Kp   = (_Float16*)(ws + off); off += (size_t)Bn * Tn * 512 * 2;    // 8 MB
  _Float16* Vp   = (_Float16*)(ws + off); off += (size_t)Bn * Tn * 512 * 2;    // 8 MB
  _Float16* Kh   = (_Float16*)(ws + off); off += (size_t)Bn * Tn * 512 * 2;    // 8 MB
  _Float16* Vt   = (_Float16*)(ws + off); off += (size_t)Bn * Tn * 512 * 2;    // 8 MB
  _Float16* Oh   = (_Float16*)(ws + off); off += (size_t)Bn * Tn * En * 2;     // 32 MB

  const int M = Bn * Tn;   // 8192

  // 1) weight transposes (f32 -> f16, K x N -> N x K)
  wtrans_kernel<<<(En * En + 255) / 256, 256, 0, stream>>>(Wq, Wq_t, En, En);
  wtrans_kernel<<<(En * 512 + 255) / 256, 256, 0, stream>>>(Wk, Wk_t, En, 512);
  wtrans_kernel<<<(En * 512 + 255) / 256, 256, 0, stream>>>(Wv, Wv_t, En, 512);
  wtrans_kernel<<<(En * En + 255) / 256, 256, 0, stream>>>(Wo, Wo_t, En, En);

  // 2) projections: x @ W + b   (fp32 A, f16 Bt, f16 out)
  gemm_wmma_kernel<float, false><<<(M / 32) * (En / 32) / 8, 256, 0, stream>>>(
      x, Wq_t, bq, (void*)Qp, M, En, En);
  gemm_wmma_kernel<float, false><<<(M / 32) * (512 / 32) / 8, 256, 0, stream>>>(
      x, Wk_t, bk, (void*)Kp, M, 512, En);
  gemm_wmma_kernel<float, false><<<(M / 32) * (512 / 32) / 8, 256, 0, stream>>>(
      x, Wv_t, bv, (void*)Vp, M, 512, En);

  // 3) RoPE on K and V (+ layout rearrange, V transposed)
  {
    long total = (long)Bn * NKVn * Tn * (HDn / 2);
    rope_kv_kernel<<<(int)((total + 255) / 256), 256, 0, stream>>>(Kp, Vp, Kh, Vt);
  }

  // 4) attention: one block per (b, h, 64-row q tile)
  attn_kernel<<<Bn * NHn * (Tn / 64), 128, 0, stream>>>(Qp, Kh, Vt, mask, Oh);

  // 5) output projection: Oh @ Wo + bo -> fp32 d_out
  gemm_wmma_kernel<_Float16, true><<<(M / 32) * (En / 32) / 8, 256, 0, stream>>>(
      Oh, Wo_t, bo, d_out, M, En, En);
}
